// CrissCrossAttention_13400297963998
// MI455X (gfx1250) — matrix operand order repaired and checked by reading, not compiled, not run on hardware
//
#include <hip/hip_runtime.h>
#include <hip/hip_bf16.h>

// CDNA5 / gfx1250, wave32. All GEMMs via v_wmma_f32_16x16x32_bf16.
// LDS tile layouts chosen so every WMMA fragment is a contiguous per-lane
// run (A: two 16B chunks, B^T: one 32B chunk) -> ds_load_b128.

typedef __attribute__((ext_vector_type(16))) __bf16 v16bf;
typedef __attribute__((ext_vector_type(8)))  __bf16 v8bf;
typedef __attribute__((ext_vector_type(8)))  float  v8f;
typedef __attribute__((ext_vector_type(4)))  int    v4i;

#define B_  8
#define C_  64
#define H_  128
#define W_  128
#define CP_ 72    // 64 + 8 pad  (row stride 144B: 16B-aligned, bank-spread)
#define GP_ 136   // 128 + 8 pad (row stride 272B: 16B-aligned, bank-spread)

// ---- optional CDNA5 async global->LDS copy (ASYNCcnt-tracked) ----
#if defined(__gfx1250__) && __has_builtin(__builtin_amdgcn_global_load_async_to_lds_b128)
#define ASYNC_LDS 1
typedef __attribute__((address_space(1))) v4i gv4i;   // global int4
typedef __attribute__((address_space(3))) v4i lv4i;   // LDS int4
__device__ __forceinline__ void async_cp16(const void* gsrc, void* ldst) {
  __builtin_amdgcn_global_load_async_to_lds_b128(
      (gv4i*)(void*)gsrc, (lv4i*)ldst, 0, 0);
}
__device__ __forceinline__ void async_wait0() {
#if __has_builtin(__builtin_amdgcn_s_wait_asynccnt)
  __builtin_amdgcn_s_wait_asynccnt(0);
#else
  asm volatile("s_wait_asynccnt 0x0" ::: "memory");
#endif
}
#endif

__device__ __forceinline__ v8f wmma_bf16(v16bf a, v16bf b, v8f c) {
  return __builtin_amdgcn_wmma_f32_16x16x32_bf16(
      false, a, false, b, (short)0, c, false, false);
}

// A (16x32 bf16, §7.12.2): lane L -> row M=L&15, half=L>>4.
// element e: K = 16*half + e (this code's order, B matches)  => one contiguous 16-elem run.
__device__ __forceinline__ v16bf load_a_frag(const __bf16* tile, int stride, int lane) {
  const __bf16* p = tile + (lane & 15) * stride + ((lane >> 4) << 4);
  v16bf f;
  #pragma unroll
  for (int e = 0; e < 8; ++e)  f[e] = p[e];
  #pragma unroll
  for (int e = 8; e < 16; ++e) f[e] = p[e];       // K = 16*half + e
  return f;
}
// B (32x16 bf16): lane L -> col N=L&15, K = 16*(L>>4) + e. Tile stored
// TRANSPOSED as [N][K] so the 16 elements are one contiguous 32B run.
__device__ __forceinline__ v16bf load_bT_frag(const __bf16* tileT, int stride, int lane) {
  const __bf16* p = tileT + (lane & 15) * stride + ((lane >> 4) << 4);
  v16bf f;
  #pragma unroll
  for (int e = 0; e < 16; ++e) f[e] = p[e];
  return f;
}
// C/D (16x16 f32, 8 VGPRs): VGPR i, lane L: M = i + 8*(L>>4), N = L&15
__device__ __forceinline__ void cd_idx(int lane, int i, int& m, int& n) {
  m = i + ((lane >> 4) << 3);
  n = lane & 15;
}
__device__ __forceinline__ v8f zero8() {
  v8f z = {0.f,0.f,0.f,0.f,0.f,0.f,0.f,0.f};
  return z;
}

// =====================================================================
// Kernel 1: q/k/v 3x3 conv as implicit GEMM. Grid (H, B, 3), block 256.
// Per tap: (64x64 weights) @ (64x128 shifted input row), K in 2x32.
// =====================================================================
__global__ __launch_bounds__(256) void k_conv_qkv(
    const float* __restrict__ x,
    const float* __restrict__ Wq, const float* __restrict__ bq,
    const float* __restrict__ Wk, const float* __restrict__ bk,
    const float* __restrict__ Wv, const float* __restrict__ bv,
    __bf16* __restrict__ qo, __bf16* __restrict__ ko, __bf16* __restrict__ vo)
{
  const int h     = blockIdx.x;
  const int b     = blockIdx.y;
  const int which = blockIdx.z;
  const float* Wc = (which == 0) ? Wq : (which == 1) ? Wk : Wv;
  const float* bc = (which == 0) ? bq : (which == 1) ? bk : bv;
  __bf16* outp    = (which == 0) ? qo : (which == 1) ? ko : vo;

  __shared__ __bf16 xsT[3][W_][CP_];   // [row][w][c]  (B^T layout) 55.3 KB
  __shared__ __bf16 wsh[C_][CP_];      // [co][ci]     (A layout)    9.2 KB
  __shared__ float  bias_s[C_];

  const int tid = threadIdx.x;

  for (int r = 0; r < 3; ++r) {
    int hh = h + r - 1;
    if (hh < 0 || hh >= H_) continue;           // uniform; skipped rows unread
    for (int i = tid; i < (C_ * W_) / 4; i += 256) {
      int c = i >> 5, w4 = (i & 31) * 4;        // float4-vectorized global reads
      float4 xv = *(const float4*)(x + (((size_t)b * C_ + c) * H_ + hh) * W_ + w4);
      xsT[r][w4 + 0][c] = (__bf16)xv.x;
      xsT[r][w4 + 1][c] = (__bf16)xv.y;
      xsT[r][w4 + 2][c] = (__bf16)xv.z;
      xsT[r][w4 + 3][c] = (__bf16)xv.w;
    }
  }
  if (tid < C_) bias_s[tid] = bc[tid];

  const int wave = tid >> 5, lane = tid & 31;
  const int w0 = wave * 16;
  const int nn = lane & 15, half = lane >> 4;

  v8f acc[4];
  for (int ct = 0; ct < 4; ++ct) acc[ct] = zero8();

  for (int t = 0; t < 9; ++t) {
    const int kh = t / 3, kw = t % 3;
    const int hh = h + kh - 1;
    __syncthreads();
    for (int i = tid; i < C_ * C_; i += 256) {
      int co = i >> 6, ci = i & 63;             // stride-9 gather, L2-hot
      wsh[co][ci] = (__bf16)Wc[((co * C_ + ci) * 3 + kh) * 3 + kw];
    }
    __syncthreads();
    if (hh < 0 || hh >= H_) continue;           // zero-padded row contributes 0

    const int wsrc = w0 + nn + kw - 1;          // per-lane, uniform over elems
    const bool okw = (wsrc >= 0 && wsrc < W_);
    v16bf bfr[2];
    for (int kc = 0; kc < 2; ++kc) {
      if (okw) {
        const __bf16* p = &xsT[kh][wsrc][kc * 32 + half * 16];
        #pragma unroll
        for (int e = 0; e < 16; ++e) bfr[kc][e] = p[e];
      } else {
        #pragma unroll
        for (int e = 0; e < 16; ++e) bfr[kc][e] = (__bf16)0.f;
      }
    }
    for (int kc = 0; kc < 2; ++kc)              // kc outer: consecutive WMMAs
      for (int ct = 0; ct < 4; ++ct)            // share the B operand
        acc[ct] = wmma_bf16(load_a_frag(&wsh[ct * 16][kc * 32], CP_, lane),
                            bfr[kc], acc[ct]);
  }

  for (int ct = 0; ct < 4; ++ct) {
    #pragma unroll
    for (int i = 0; i < 8; ++i) {
      int m, n; cd_idx(lane, i, m, n);
      int c = ct * 16 + m, w = w0 + n;
      outp[(((size_t)b * C_ + c) * H_ + h) * W_ + w] =
          (__bf16)(acc[ct][i] + bias_s[c]);
    }
  }
}

// =====================================================================
// Kernel 2: eH[b,h,w,g] = sum_c q[c,h,w]*k[c,g,w], diag masked. Grid (W,B).
// =====================================================================
__global__ __launch_bounds__(256) void k_energy_col(
    const __bf16* __restrict__ q, const __bf16* __restrict__ k,
    __bf16* __restrict__ att)
{
  const int w = blockIdx.x, b = blockIdx.y;
  __shared__ __bf16 QsT[H_][CP_];   // [h][c]  A layout
  __shared__ __bf16 KsT[H_][CP_];   // [g][c]  B^T layout
  const int tid = threadIdx.x;
  for (int i = tid; i < C_ * H_; i += 256) {    // inherent column gather
    int c = i >> 7, hg = i & 127;
    size_t idx = (((size_t)b * C_ + c) * H_ + hg) * W_ + w;
    QsT[hg][c] = q[idx];
    KsT[hg][c] = k[idx];
  }
  __syncthreads();
  const int wave = tid >> 5, lane = tid & 31;
  const int h0 = wave * 16;
  for (int gt = 0; gt < 8; ++gt) {
    v8f acc = zero8();
    for (int kc = 0; kc < 2; ++kc)
      acc = wmma_bf16(load_a_frag(&QsT[h0][kc * 32], CP_, lane),
                      load_bT_frag(&KsT[gt * 16][kc * 32], CP_, lane), acc);
    #pragma unroll
    for (int i = 0; i < 8; ++i) {
      int m, n; cd_idx(lane, i, m, n);
      int hh = h0 + m, g = gt * 16 + n;
      float val = (g == hh) ? -3.0e38f : acc[i];
      att[((((size_t)b * H_ + hh) * W_ + w) << 8) + g] = (__bf16)val;
    }
  }
}

// =====================================================================
// Kernel 3: eW[b,h,w,g] = sum_c q[c,h,w]*k[c,h,g]. Grid (H,B).
// =====================================================================
__global__ __launch_bounds__(256) void k_energy_row(
    const __bf16* __restrict__ q, const __bf16* __restrict__ k,
    __bf16* __restrict__ att)
{
  const int h = blockIdx.x, b = blockIdx.y;
  __shared__ __bf16 QsT[W_][CP_];   // [w][c]
  __shared__ __bf16 KsT[W_][CP_];   // [g][c]
  const int tid = threadIdx.x;
  for (int i = tid; i < (C_ * W_) / 8; i += 256) {
    int c = i >> 4, wg8 = (i & 15) * 8;         // v8bf global loads, LDS scatter
    size_t idx = (((size_t)b * C_ + c) * H_ + h) * W_ + wg8;
    v8bf qv = *(const v8bf*)(q + idx);
    v8bf kv = *(const v8bf*)(k + idx);
    #pragma unroll
    for (int e = 0; e < 8; ++e) {
      QsT[wg8 + e][c] = qv[e];
      KsT[wg8 + e][c] = kv[e];
    }
  }
  __syncthreads();
  const int wave = tid >> 5, lane = tid & 31;
  const int w0 = wave * 16;
  for (int gt = 0; gt < 8; ++gt) {
    v8f acc = zero8();
    for (int kc = 0; kc < 2; ++kc)
      acc = wmma_bf16(load_a_frag(&QsT[w0][kc * 32], CP_, lane),
                      load_bT_frag(&KsT[gt * 16][kc * 32], CP_, lane), acc);
    #pragma unroll
    for (int i = 0; i < 8; ++i) {
      int m, n; cd_idx(lane, i, m, n);
      att[((((size_t)b * H_ + h) * W_ + (w0 + m)) << 8) + 128 + (gt * 16 + n)] =
          (__bf16)acc[i];
    }
  }
}

// =====================================================================
// Kernel 4: per-position softmax stats (max, 1/sum). One wave32 per
// position; each lane loads its 16B chunk (order-insensitive reduce).
// =====================================================================
__global__ __launch_bounds__(256) void k_reduce(
    const __bf16* __restrict__ att, float2* __restrict__ stats)
{
  const int wave = threadIdx.x >> 5, lane = threadIdx.x & 31;
  const size_t pos = (size_t)blockIdx.x * 8 + wave;
  v8bf ch = *(const v8bf*)(att + pos * 256 + (size_t)lane * 8);  // b128/lane
  float vals[8];
  float mx = -3.4e38f;
  #pragma unroll
  for (int i = 0; i < 8; ++i) {
    vals[i] = (float)ch[i];
    mx = fmaxf(mx, vals[i]);
  }
  #pragma unroll
  for (int off = 16; off > 0; off >>= 1) mx = fmaxf(mx, __shfl_xor(mx, off, 32));
  float sum = 0.f;
  #pragma unroll
  for (int i = 0; i < 8; ++i) sum += __expf(vals[i] - mx);
  #pragma unroll
  for (int off = 16; off > 0; off >>= 1) sum += __shfl_xor(sum, off, 32);
  if (lane == 0) stats[pos] = make_float2(mx, 1.f / sum);
}

// =====================================================================
// Kernel 5a: out = gamma * (Vrow @ softmax(AttW)^T) + x   (per (b,h))
// =====================================================================
__global__ __launch_bounds__(256) void k_out_row(
    const __bf16* __restrict__ v, const __bf16* __restrict__ att,
    const float2* __restrict__ stats, const float* __restrict__ x,
    const float* __restrict__ gamma, float* __restrict__ out)
{
  const int h = blockIdx.x, b = blockIdx.y;
  __shared__ __bf16 Vs[C_][GP_];    // [c][g]  A layout (17.4 KB)
  __shared__ __bf16 Bt[W_][GP_];    // [w][g]  B^T layout = natural att order
  const int tid = threadIdx.x;

  // V tile: contiguous bf16 copy, 16B chunks.
#ifdef ASYNC_LDS
  for (int j = tid; j < (C_ * W_) / 8; j += 256) {
    int c = j >> 4, g8 = (j & 15) * 8;
    async_cp16(v + ((((size_t)b * C_ + c) * H_ + h) * W_) + g8, &Vs[c][g8]);
  }
#else
  for (int j = tid; j < (C_ * W_) / 8; j += 256) {
    int c = j >> 4, g8 = (j & 15) * 8;
    *(v8bf*)&Vs[c][g8] =
        *(const v8bf*)(v + ((((size_t)b * C_ + c) * H_ + h) * W_) + g8);
  }
#endif
  // B tile: apply softmax on the fly (overlaps with async V copy above).
  for (int i = tid; i < (W_ * W_) / 8; i += 256) {
    int w = i >> 4, g8 = (i & 15) * 8;
    size_t pbase = ((size_t)b * H_ + h) * W_ + w;
    float2 s = stats[pbase];
    v8bf lv = *(const v8bf*)(att + (pbase << 8) + 128 + g8);
    v8bf o;
    #pragma unroll
    for (int e = 0; e < 8; ++e)
      o[e] = (__bf16)(__expf((float)lv[e] - s.x) * s.y);
    *(v8bf*)&Bt[w][g8] = o;
  }
#ifdef ASYNC_LDS
  async_wait0();
#endif
  __syncthreads();

  const float gm = gamma[0];
  const int wave = tid >> 5, lane = tid & 31;
  const int w0 = wave * 16;
  for (int ct = 0; ct < 4; ++ct) {
    v8f acc = zero8();
    for (int kc = 0; kc < 4; ++kc)   // K = 128 = 4 x 32
      acc = wmma_bf16(load_a_frag(&Vs[ct * 16][kc * 32], GP_, lane),
                      load_bT_frag(&Bt[w0][kc * 32], GP_, lane), acc);
    #pragma unroll
    for (int i = 0; i < 8; ++i) {
      int m, n; cd_idx(lane, i, m, n);
      int c = ct * 16 + m, w = w0 + n;
      size_t idx = (((size_t)b * C_ + c) * H_ + h) * W_ + w;
      out[idx] = gm * acc[i] + x[idx];
    }
  }
}

// =====================================================================
// Kernel 5b: out += gamma * (Vcol @ softmax(AttH)^T)  (per (b,w); runs
// after 5a fully wrote out -> stream-ordered RMW is race-free)
// =====================================================================
__global__ __launch_bounds__(256) void k_out_col(
    const __bf16* __restrict__ v, const __bf16* __restrict__ att,
    const float2* __restrict__ stats, const float* __restrict__ gamma,
    float* __restrict__ out)
{
  const int w = blockIdx.x, b = blockIdx.y;
  __shared__ __bf16 Vs[C_][GP_];    // [c][g] = v[b,c,g,w]
  __shared__ __bf16 Bt[H_][GP_];    // [h][g] = softmaxed attH
  const int tid = threadIdx.x;
  for (int i = tid; i < C_ * H_; i += 256) {    // inherent column gather
    int c = i >> 7, g = i & 127;
    Vs[c][g] = v[(((size_t)b * C_ + c) * H_ + g) * W_ + w];
  }
  for (int i = tid; i < (H_ * H_) / 8; i += 256) {
    int hh = i >> 4, g8 = (i & 15) * 8;
    size_t pbase = ((size_t)b * H_ + hh) * W_ + w;
    float2 s = stats[pbase];
    v8bf lv = *(const v8bf*)(att + (pbase << 8) + g8);
    v8bf o;
    #pragma unroll
    for (int e = 0; e < 8; ++e)
      o[e] = (__bf16)(__expf((float)lv[e] - s.x) * s.y);
    *(v8bf*)&Bt[hh][g8] = o;
  }
  __syncthreads();
  const float gm = gamma[0];
  const int wave = tid >> 5, lane = tid & 31;
  const int h0 = wave * 16;
  for (int ct = 0; ct < 4; ++ct) {
    v8f acc = zero8();
    for (int kc = 0; kc < 4; ++kc)
      acc = wmma_bf16(load_a_frag(&Vs[ct * 16][kc * 32], GP_, lane),
                      load_bT_frag(&Bt[h0][kc * 32], GP_, lane), acc);
    #pragma unroll
    for (int i = 0; i < 8; ++i) {
      int m, n; cd_idx(lane, i, m, n);
      int c = ct * 16 + m, hh = h0 + n;
      size_t idx = (((size_t)b * C_ + c) * H_ + hh) * W_ + w;
      out[idx] += gm * acc[i];
    }
  }
}

// =====================================================================
extern "C" void kernel_launch(void* const* d_in, const int* in_sizes, int n_in,
                              void* d_out, int out_size, void* d_ws, size_t ws_size,
                              hipStream_t stream) {
  const float* x     = (const float*)d_in[0];
  const float* Wq    = (const float*)d_in[1];
  const float* bq    = (const float*)d_in[2];
  const float* Wk    = (const float*)d_in[3];
  const float* bk    = (const float*)d_in[4];
  const float* Wv    = (const float*)d_in[5];
  const float* bv    = (const float*)d_in[6];
  const float* gamma = (const float*)d_in[7];
  float* out = (float*)d_out;

  const size_t nchw = (size_t)B_ * C_ * H_ * W_;      // 8,388,608
  const size_t npos = (size_t)B_ * H_ * W_;           // 131,072
  __bf16* qb    = (__bf16*)d_ws;
  __bf16* kb    = qb + nchw;
  __bf16* vb    = kb + nchw;
  __bf16* att   = vb + nchw;          // [B,H,W,256] bf16 = 67.1 MB
  float2* stats = (float2*)(att + npos * 256);   // 1 MB; ws total ~113 MB

  dim3 blk(256);
  k_conv_qkv  <<<dim3(H_, B_, 3), blk, 0, stream>>>(x, Wq, bq, Wk, bk, Wv, bv,
                                                    qb, kb, vb);
  k_energy_col<<<dim3(W_, B_), blk, 0, stream>>>(qb, kb, att);
  k_energy_row<<<dim3(H_, B_), blk, 0, stream>>>(qb, kb, att);
  k_reduce    <<<dim3((unsigned)(npos / 8)), blk, 0, stream>>>(att, stats);
  k_out_row   <<<dim3(H_, B_), blk, 0, stream>>>(vb, att, stats, x, gamma, out);
  k_out_col   <<<dim3(W_, B_), blk, 0, stream>>>(vb, att, stats, gamma, out);
}
